// AttentionGRU_47605417508938
// MI455X (gfx1250) — matrix operand order repaired and checked by reading, not compiled, not run on hardware
//
#include <hip/hip_runtime.h>
#include <hip/hip_bf16.h>

// ---------------------------------------------------------------------------
// AttentionGRU for MI455X (gfx1250, wave32, WMMA bf16 16x16x32)
//   Phase 1: persistent cooperative GRU scan (32 WGs, global spin barrier),
//            x-proj and h-proj GEMMs fused per step, weights resident in LDS.
//   Phase 2: attention algebraically collapsed -> no k/v materialization.
// ---------------------------------------------------------------------------

typedef __attribute__((ext_vector_type(16))) __bf16 v16bf;
typedef __attribute__((ext_vector_type(8)))  __bf16 v8bf;
typedef __attribute__((ext_vector_type(8)))  float  v8f;

#define L_SEQ   512
#define BATCH   128
#define DIMH    512
#define NWG     32        // scan workgroups, 16 hidden cols each
#define WROW    520       // padded K-length for weight LDS rows (bank spread)
#define XROW    136       // padded K-length for A/B staging rows

// ---------------- WMMA fragment helpers (ISA 7.12.2 layouts) ----------------

__device__ __forceinline__ v8f wmma_bf16(v16bf a, v16bf b, v8f c) {
    return __builtin_amdgcn_wmma_f32_16x16x32_bf16(
        /*neg_a=*/false, a, /*neg_b=*/false, b,
        /*c_mod=*/(short)0, c, /*reuse_a=*/false, /*reuse_b=*/false);
}

// A: 16x32 (MxK). lane<16 holds K {0..7,16..23}, lane>=16 holds K {8..15,24..31}.
__device__ __forceinline__ v16bf load_a_frag(const __bf16* tile, int ld,
                                             int rowBase, int k0) {
    const int l = threadIdx.x & 31;
    const __bf16* p = tile + (rowBase + (l & 15)) * ld + k0 + ((l & 16) ? 8 : 0);
    v8bf lo = *(const v8bf*)(p);
    v8bf hi = *(const v8bf*)(p + 16);
    v16bf a;
#pragma unroll
    for (int e = 0; e < 8; ++e) { a[e] = lo[e]; a[e + 8] = hi[e]; }
    return a;
}

// B: 32x16 (KxN), stored in LDS as [n][k]. lane<16 holds K {0..7,16..23}, lane>=16 holds K {8..15,24..31}.
__device__ __forceinline__ v16bf load_b_frag(const __bf16* tile, int ld, int k0) {
    const int l = threadIdx.x & 31;
    const __bf16* p = tile + (l & 15) * ld + k0 + ((l & 16) ? 8 : 0);
    v8bf lo = *(const v8bf*)(p);
    v8bf hi = *(const v8bf*)(p + 16);
    v16bf b;
#pragma unroll
    for (int e = 0; e < 8; ++e) { b[e] = lo[e]; b[e + 8] = hi[e]; }
    return b;
}

__device__ __forceinline__ float sigmoidf(float x) {
    return 1.0f / (1.0f + __expf(-x));
}

__device__ __forceinline__ void global_barrier(unsigned* bar, unsigned target) {
    __syncthreads();
    if (threadIdx.x == 0) {
        __threadfence();
        atomicAdd(bar, 1u);
        while (__atomic_load_n(bar, __ATOMIC_ACQUIRE) < target) {
            __builtin_amdgcn_s_sleep(2);
        }
    }
    __syncthreads();
}

// ------------------------------ init -----------------------------------

__global__ void init_kernel(unsigned* bar) {
    if (threadIdx.x == 0) *bar = 0u;
}

// --------------------------- GRU scan ----------------------------------
// grid = 32 blocks x 256 threads (8 waves). Block g owns hidden cols
// [16g,16g+16). Wave w owns batch rows [16w,16w+16). Per step: fused
// x@W_ih^T + h@W_hh^T for this WG's 3 gate column-tiles, gate math,
// write h (f32 state + bf16 operand copy, double buffered), global barrier.

__global__ __launch_bounds__(256) void gru_scan_kernel(
    const float* __restrict__ x,
    const float* __restrict__ w_ih, const float* __restrict__ w_hh,
    const float* __restrict__ b_ih, const float* __restrict__ b_hh,
    float* __restrict__ h_f32, __bf16* __restrict__ h_bf,
    unsigned* __restrict__ bar)
{
    // weights: [mat(ih/hh)][gate][n][k] bf16, resident whole scan (97.5 KB)
    __shared__ __align__(32) __bf16 wT[2][3 * 16 * WROW];
    __shared__ __align__(32) __bf16 xs[BATCH * XROW];   // x[t] K-chunk (34 KB)
    __shared__ __align__(32) __bf16 hs[BATCH * XROW];   // h    K-chunk (34 KB)

    const int tid  = threadIdx.x;
    const int wave = tid >> 5;
    const int lane = tid & 31;
    const int c0   = blockIdx.x * 16;     // hidden column base for this WG
    const int m0   = wave * 16;           // batch row base for this wave

    // Preload this WG's weight columns as bf16 in [n][k] (B-fragment) layout.
    for (int idx = tid; idx < 3 * 16 * 512; idx += 256) {
        const int k = idx & 511;
        const int n = (idx >> 9) & 15;
        const int g = idx >> 13;
        wT[0][(g * 16 + n) * WROW + k] = (__bf16)w_ih[(g * 512 + c0 + n) * 512 + k];
        wT[1][(g * 16 + n) * WROW + k] = (__bf16)w_hh[(g * 512 + c0 + n) * 512 + k];
    }
    // h0 = 0: each WG zeroes its own columns (f32 state + bf16 buffer 0).
    for (int idx = tid; idx < BATCH * 16; idx += 256) {
        const int m = idx >> 4, n = idx & 15;
        h_f32[m * DIMH + c0 + n] = 0.0f;
        h_bf[m * DIMH + c0 + n]  = (__bf16)0.0f;
    }
    // Per-lane gate biases for this lane's output column.
    const int   mycol = c0 + (lane & 15);
    const float br  = b_ih[mycol] + b_hh[mycol];
    const float bz  = b_ih[512 + mycol] + b_hh[512 + mycol];
    const float bxn = b_ih[1024 + mycol];
    const float bhn = b_hh[1024 + mycol];

    unsigned tgt = NWG;
    global_barrier(bar, tgt);             // h init visible everywhere

    for (int t = 0; t < L_SEQ; ++t) {
        const __bf16* hcur  = h_bf + (size_t)(t & 1) * BATCH * DIMH;
        __bf16*       hnext = h_bf + (size_t)((t + 1) & 1) * BATCH * DIMH;
        const float*  xt    = x + (size_t)t * BATCH * 512;

        v8f acc_r = {}, acc_z = {}, acc_xn = {}, acc_hn = {};

        for (int kc = 0; kc < 4; ++kc) {
            const int kg = kc * 128;
            for (int idx = tid; idx < BATCH * 128; idx += 256) {
                const int m = idx >> 7, k = idx & 127;
                xs[m * XROW + k] = (__bf16)xt[m * 512 + kg + k];
                hs[m * XROW + k] = hcur[m * 512 + kg + k];
            }
            __syncthreads();
#pragma unroll
            for (int ks = 0; ks < 4; ++ks) {
                const int k0 = ks * 32;
                const int kk = kg + k0;
                // Issue ALL fragment loads for this k-step before any WMMA so
                // the six WMMAs drain back-to-back behind a single dscnt wait.
                v16bf ax  = load_a_frag(xs, XROW, m0, k0);
                v16bf ah  = load_a_frag(hs, XROW, m0, k0);
                v16bf bir = load_b_frag(&wT[0][0 * 16 * WROW], WROW, kk);
                v16bf bhr = load_b_frag(&wT[1][0 * 16 * WROW], WROW, kk);
                v16bf biz = load_b_frag(&wT[0][1 * 16 * WROW], WROW, kk);
                v16bf bhz = load_b_frag(&wT[1][1 * 16 * WROW], WROW, kk);
                v16bf bin = load_b_frag(&wT[0][2 * 16 * WROW], WROW, kk);
                v16bf bhn_ = load_b_frag(&wT[1][2 * 16 * WROW], WROW, kk);
                acc_r  = wmma_bf16(ax, bir, acc_r);
                acc_r  = wmma_bf16(ah, bhr, acc_r);
                acc_z  = wmma_bf16(ax, biz, acc_z);
                acc_z  = wmma_bf16(ah, bhz, acc_z);
                acc_xn = wmma_bf16(ax, bin, acc_xn);
                acc_hn = wmma_bf16(ah, bhn_, acc_hn);
            }
            __syncthreads();
        }

        // Gate nonlinearity + state update (C-layout: VGPR j -> rows j, j+8).
#pragma unroll
        for (int j = 0; j < 8; ++j) {
            const int m = m0 + j + ((lane & 16) ? 8 : 0);
            const float r = sigmoidf(acc_r[j] + br);
            const float z = sigmoidf(acc_z[j] + bz);
            const float n = tanhf(acc_xn[j] + bxn + r * (acc_hn[j] + bhn));
            const float hold = h_f32[m * DIMH + mycol];
            const float hnew = (1.0f - z) * n + z * hold;
            h_f32[m * DIMH + mycol] = hnew;
            hnext[m * DIMH + mycol] = (__bf16)hnew;
        }
        tgt += NWG;
        global_barrier(bar, tgt);
    }
}

// ------------------ generic strided WMMA GEMM ---------------------------
// C[m][n] = sum_k A(m,k)*B(k,n) + bias[n] + Cin[m][n]
// A(m,k)=A[m*sAm + k*sAk], B(k,n)=B[k*sBk + n*sBn]. grid=(M/128, N/16).

__global__ __launch_bounds__(256) void wmma_gemm_kernel(
    const float* __restrict__ A, int sAm, int sAk,
    const float* __restrict__ B, int sBk, int sBn,
    const float* __restrict__ bias, const float* __restrict__ Cin,
    float* __restrict__ C, int N, int K)
{
    __shared__ __align__(32) __bf16 As[128 * XROW];
    __shared__ __align__(32) __bf16 Bs[16 * XROW];
    const int tid = threadIdx.x, wave = tid >> 5, lane = tid & 31;
    const int mBase = blockIdx.x * 128;
    const int nBase = blockIdx.y * 16;

    v8f acc = {};
    for (int kc = 0; kc < K; kc += 128) {
        for (int idx = tid; idx < 128 * 128; idx += 256) {
            const int m = idx >> 7, k = idx & 127;
            As[m * XROW + k] =
                (__bf16)A[(size_t)(mBase + m) * sAm + (size_t)(kc + k) * sAk];
        }
        for (int idx = tid; idx < 16 * 128; idx += 256) {
            const int n = idx >> 7, k = idx & 127;
            Bs[n * XROW + k] =
                (__bf16)B[(size_t)(kc + k) * sBk + (size_t)(nBase + n) * sBn];
        }
        __syncthreads();
        // Load all fragments first, then issue the four WMMAs together.
        v16bf a0 = load_a_frag(As, XROW, wave * 16, 0);
        v16bf b0 = load_b_frag(Bs, XROW, 0);
        v16bf a1 = load_a_frag(As, XROW, wave * 16, 32);
        v16bf b1 = load_b_frag(Bs, XROW, 32);
        v16bf a2 = load_a_frag(As, XROW, wave * 16, 64);
        v16bf b2 = load_b_frag(Bs, XROW, 64);
        v16bf a3 = load_a_frag(As, XROW, wave * 16, 96);
        v16bf b3 = load_b_frag(Bs, XROW, 96);
        acc = wmma_bf16(a0, b0, acc);
        acc = wmma_bf16(a1, b1, acc);
        acc = wmma_bf16(a2, b2, acc);
        acc = wmma_bf16(a3, b3, acc);
        __syncthreads();
    }
    const int n = nBase + (lane & 15);
    const float bb = bias ? bias[n] : 0.0f;
#pragma unroll
    for (int j = 0; j < 8; ++j) {
        const int m = mBase + wave * 16 + j + ((lane & 16) ? 8 : 0);
        float v = acc[j] + bb;
        if (Cin) v += Cin[(size_t)m * N + n];
        C[(size_t)m * N + n] = v;
    }
}

// -------- scores[l,b] = x[l,b,:].qk[b,:] + spos[l,b] (one wave per pair) ----

__global__ __launch_bounds__(256) void scores_kernel(
    const float* __restrict__ x, const float* __restrict__ qk,
    const float* __restrict__ spos, float* __restrict__ scores)
{
    const int wave = threadIdx.x >> 5, lane = threadIdx.x & 31;
    const int p = blockIdx.x * 8 + wave;        // p = l*128 + b
    const int b = p & 127;
    const float* xr = x + (size_t)p * 512;
    const float* qr = qk + (size_t)b * 512;
    float s = 0.0f;
#pragma unroll 4
    for (int j = lane; j < 512; j += 32) s += xr[j] * qr[j];
#pragma unroll
    for (int off = 16; off; off >>= 1) s += __shfl_xor(s, off, 32);
    if (lane == 0) scores[p] = s + spos[p];
}

// ---------------- softmax over l, per batch column (in-place) -------------

__global__ __launch_bounds__(256) void softmax_kernel(float* __restrict__ sc) {
    __shared__ float red[256];
    const int b = blockIdx.x, tid = threadIdx.x;
    const float v0 = sc[tid * 128 + b];
    const float v1 = sc[(tid + 256) * 128 + b];
    red[tid] = fmaxf(v0, v1);
    __syncthreads();
    for (int s = 128; s; s >>= 1) {
        if (tid < s) red[tid] = fmaxf(red[tid], red[tid + s]);
        __syncthreads();
    }
    const float mx = red[0];
    __syncthreads();
    const float e0 = __expf(v0 - mx), e1 = __expf(v1 - mx);
    red[tid] = e0 + e1;
    __syncthreads();
    for (int s = 128; s; s >>= 1) {
        if (tid < s) red[tid] += red[tid + s];
        __syncthreads();
    }
    const float inv = 1.0f / red[0];
    sc[tid * 128 + b] = e0 * inv;
    sc[(tid + 256) * 128 + b] = e1 * inv;
}

// ------------- wx[b,j] = sum_l a[l,b] * x[l,b,j] (one block per b) ----------

__global__ __launch_bounds__(256) void wsum_kernel(
    const float* __restrict__ x, const float* __restrict__ a,
    float* __restrict__ wx)
{
    const int b = blockIdx.x, tid = threadIdx.x;
    float a0 = 0.0f, a1 = 0.0f;
    for (int l = 0; l < 512; ++l) {
        const float al = a[l * 128 + b];
        const float* xr = x + ((size_t)l * 128 + b) * 512;
        a0 += al * xr[tid];
        a1 += al * xr[tid + 256];
    }
    wx[(size_t)b * 512 + tid] = a0;
    wx[(size_t)b * 512 + tid + 256] = a1;
}

// ------------------------------- launcher ---------------------------------

extern "C" void kernel_launch(void* const* d_in, const int* in_sizes, int n_in,
                              void* d_out, int out_size, void* d_ws, size_t ws_size,
                              hipStream_t stream)
{
    (void)in_sizes; (void)n_in; (void)out_size; (void)ws_size;
    const float* x    = (const float*)d_in[0];
    const float* w_ih = (const float*)d_in[1];
    const float* w_hh = (const float*)d_in[2];
    const float* b_ih = (const float*)d_in[3];
    const float* b_hh = (const float*)d_in[4];
    const float* Qw   = (const float*)d_in[5];
    const float* Qb   = (const float*)d_in[6];
    const float* Kw   = (const float*)d_in[7];
    // d_in[8] = Kb: adds a constant to every score row -> softmax-invariant.
    const float* Vw   = (const float*)d_in[9];
    const float* Vb   = (const float*)d_in[10];
    const float* pos  = (const float*)d_in[11];
    float* out = (float*)d_out;

    char* p = (char*)d_ws;
    unsigned* bar  = (unsigned*)p;  p += 256;
    float*   h     = (float*)p;     p += (size_t)128 * 512 * 4;   // final GRU state
    __bf16*  hbf   = (__bf16*)p;    p += (size_t)2 * 128 * 512 * 2; // double-buffer
    float*   q     = (float*)p;     p += (size_t)128 * 512 * 4;
    float*   qk    = (float*)p;     p += (size_t)128 * 512 * 4;
    float*   qp    = (float*)p;     p += (size_t)128 * 512 * 4;
    float*   spos  = (float*)p;     p += (size_t)512 * 128 * 4;
    float*   pv    = (float*)p;     p += (size_t)512 * 512 * 4;
    float*   sc    = (float*)p;     p += (size_t)512 * 128 * 4;   // scores -> a
    float*   wx    = (float*)p;     p += (size_t)128 * 512 * 4;
    float*   ap    = (float*)p;     p += (size_t)128 * 512 * 4;

    init_kernel<<<1, 1, 0, stream>>>(bar);

    // Phase 1: sequential GRU scan (dominant cost).
    gru_scan_kernel<<<NWG, 256, 0, stream>>>(x, w_ih, w_hh, b_ih, b_hh,
                                             h, hbf, bar);

    // Phase 2: collapsed attention (all tiny WMMA GEMMs + 2 streams over x).
    // q = h @ Qw^T + Qb                          (128x512, K=512)
    wmma_gemm_kernel<<<dim3(1, 32), 256, 0, stream>>>(
        h, 512, 1, Qw, 1, 512, Qb, nullptr, q, 512, 512);
    // qk = q @ Kw[:, :512]                       (128x512)
    wmma_gemm_kernel<<<dim3(1, 32), 256, 0, stream>>>(
        q, 512, 1, Kw, 1024, 1, nullptr, nullptr, qk, 512, 512);
    // qp = q @ Kw[:, 512:]                       (128x512)
    wmma_gemm_kernel<<<dim3(1, 32), 256, 0, stream>>>(
        q, 512, 1, Kw + 512, 1024, 1, nullptr, nullptr, qp, 512, 512);
    // spos = pos @ qp^T                          (512x128)
    wmma_gemm_kernel<<<dim3(4, 8), 256, 0, stream>>>(
        pos, 512, 1, qp, 1, 512, nullptr, nullptr, spos, 128, 512);
    // pv = pos @ Vw[:, 512:]^T                   (512x512)
    wmma_gemm_kernel<<<dim3(4, 32), 256, 0, stream>>>(
        pos, 512, 1, Vw + 512, 1, 1024, nullptr, nullptr, pv, 512, 512);
    // scores, softmax over l, weighted x reduction
    scores_kernel<<<8192, 256, 0, stream>>>(x, qk, spos, sc);
    softmax_kernel<<<128, 256, 0, stream>>>(sc);
    wsum_kernel<<<128, 256, 0, stream>>>(x, sc, wx);
    // ap = a^T @ pv                              (128x512)
    wmma_gemm_kernel<<<dim3(1, 32), 256, 0, stream>>>(
        sc, 1, 128, pv, 512, 1, nullptr, nullptr, ap, 512, 512);
    // out = wx @ Vw[:, :512]^T + Vb + ap         (128x512)
    wmma_gemm_kernel<<<dim3(1, 32), 256, 0, stream>>>(
        wx, 512, 1, Vw, 1, 1024, Vb, ap, out, 512, 512);
}
